// BC_51694226374867
// MI455X (gfx1250) — matrix operand order repaired and checked by reading, not compiled, not run on hardware
//
#include <hip/hip_runtime.h>
#include <hip/hip_bf16.h>
#include <cmath>

#define EMB   768
#define NSEQ  2048
#define NBAT  2
#define NHEAD 12
#define DH    64
#define TOKENS (NBAT * NSEQ)   // 4096

typedef __bf16 bf16_t;
typedef bf16_t v16bf __attribute__((ext_vector_type(16)));
typedef float  v8f   __attribute__((ext_vector_type(8)));

union Frag {
    v16bf v;
    uint4 q[2];
};

__device__ __forceinline__ v8f v8f_zero() {
    v8f z = {0.f, 0.f, 0.f, 0.f, 0.f, 0.f, 0.f, 0.f};
    return z;
}

// fp32 -> bf16 round-to-nearest-even
__device__ __forceinline__ unsigned short f2bf(float f) {
    unsigned int u = __float_as_uint(f);
    u += 0x7fffu + ((u >> 16) & 1u);
    return (unsigned short)(u >> 16);
}

__device__ __forceinline__ v8f wmma_bf16(Frag a, Frag b, v8f c) {
    return __builtin_amdgcn_wmma_f32_16x16x32_bf16(
        /*neg_a=*/false, a.v, /*neg_b=*/false, b.v,
        /*c_mod=*/(short)0, c, /*reuse_a=*/false, /*reuse_b=*/false);
}

// tanh-GELU rewritten as sigmoid: 0.5x(1+tanh(z)) = x / (1 + exp(-2z)); branch-free
__device__ __forceinline__ float gelu_fast(float x) {
    const float u = -1.5957691216057308f * (x + 0.044715f * x * x * x);
    return x * __builtin_amdgcn_rcpf(1.0f + __expf(u));
}

// ---------------------------------------------------------------------------
// GEMM: C[M,N] = act(A_bf16[M,K] @ Wt_bf16[N,K]^T + bias) + res
// Block: 256 thr (8 waves, 4x2). Tile 128x128, K-step 32 (one WMMA depth).
// Weights are pre-transposed [N][K] so both LDS tiles stage with b128 copies.
// ---------------------------------------------------------------------------
template <bool BIAS, bool RES, bool ACT, bool OUTBF>
__global__ __launch_bounds__(256) void gemm_bf16(
    const unsigned short* __restrict__ A, const unsigned short* __restrict__ Wt,
    const float* __restrict__ bias, const float* __restrict__ res,
    float* __restrict__ Cf, unsigned short* __restrict__ Cb,
    int M, int N, int K)
{
    __shared__ __align__(16) unsigned short lA[128][40];  // [m][k], pad 8
    __shared__ __align__(16) unsigned short lB[128][40];  // [n][k], pad 8

    const int t    = threadIdx.x;
    const int wave = t >> 5, lane = t & 31;
    const int ln   = lane & 15, g = lane >> 4;
    const int wrow = wave >> 1, wcol = wave & 1;   // 4 x 2 wave grid
    const int r0   = blockIdx.y * 128;
    const int c0   = blockIdx.x * 128;

    v8f acc[2][4];
    for (int mi = 0; mi < 2; ++mi)
        for (int ni = 0; ni < 4; ++ni)
            acc[mi][ni] = v8f_zero();

    // staging: 128 rows x 32 k = 4096 elems = 512 x 16B chunks, 2 per thread
    const int rr0 = t >> 2,               kk0 = (t & 3) * 8;
    const int rr1 = (t + 256) >> 2,       kk1 = kk0;

    for (int k0 = 0; k0 < K; k0 += 32) {
        uint4 a0 = *(const uint4*)(A  + (size_t)(r0 + rr0) * K + k0 + kk0);
        uint4 a1 = *(const uint4*)(A  + (size_t)(r0 + rr1) * K + k0 + kk1);
        uint4 b0 = *(const uint4*)(Wt + (size_t)(c0 + rr0) * K + k0 + kk0);
        uint4 b1 = *(const uint4*)(Wt + (size_t)(c0 + rr1) * K + k0 + kk1);
        if (k0 + 32 < K) {   // prefetch next K tile (global_prefetch_b8)
            __builtin_prefetch(A  + (size_t)(r0 + rr0) * K + k0 + 32 + kk0, 0, 0);
            __builtin_prefetch(Wt + (size_t)(c0 + rr0) * K + k0 + 32 + kk0, 0, 0);
        }

        __syncthreads();   // previous iteration's LDS reads done
        *(uint4*)&lA[rr0][kk0] = a0;
        *(uint4*)&lA[rr1][kk1] = a1;
        *(uint4*)&lB[rr0][kk0] = b0;
        *(uint4*)&lB[rr1][kk1] = b1;
        __syncthreads();

        Frag afr[2], bfr[4];
        #pragma unroll
        for (int mi = 0; mi < 2; ++mi) {
            const int rr = wrow * 32 + mi * 16 + ln;
            afr[mi].q[0] = *(const uint4*)&lA[rr][8 * g];        // K = 8g..8g+7
            afr[mi].q[1] = *(const uint4*)&lA[rr][16 + 8 * g];   // K = 16+8g..
        }
        #pragma unroll
        for (int ni = 0; ni < 4; ++ni) {
            const int cc = wcol * 64 + ni * 16 + ln;
            bfr[ni].q[0] = *(const uint4*)&lB[cc][8 * g];        // K = 8g..8g+7
            bfr[ni].q[1] = *(const uint4*)&lB[cc][16 + 8 * g];   // K = 16+8g..
        }
        #pragma unroll
        for (int mi = 0; mi < 2; ++mi)
            #pragma unroll
            for (int ni = 0; ni < 4; ++ni)
                acc[mi][ni] = wmma_bf16(afr[mi], bfr[ni], acc[mi][ni]);
    }

    // epilogue: C layout — lane holds col = ln, rows g*8 + r (branch-free)
    #pragma unroll
    for (int mi = 0; mi < 2; ++mi) {
        #pragma unroll
        for (int ni = 0; ni < 4; ++ni) {
            const int colc = c0 + wcol * 64 + ni * 16 + ln;
            const float bv = BIAS ? bias[colc] : 0.0f;
            #pragma unroll
            for (int r = 0; r < 8; ++r) {
                const int rowc = r0 + wrow * 32 + mi * 16 + g * 8 + r;
                float v = acc[mi][ni][r] + bv;
                if (ACT) v = gelu_fast(v);
                if (RES) v += res[(size_t)rowc * N + colc];
                if (OUTBF) Cb[(size_t)rowc * N + colc] = f2bf(v);
                else       Cf[(size_t)rowc * N + colc] = v;
            }
        }
    }
}

// ---------------------------------------------------------------------------
// LayerNorm over E=768; one block per token; bf16 or fp32 output
// ---------------------------------------------------------------------------
__global__ __launch_bounds__(256) void ln_kernel(
    const float* __restrict__ x, const float* __restrict__ gam,
    const float* __restrict__ bet, unsigned short* __restrict__ ybf,
    float* __restrict__ yf)
{
    __shared__ float ls[8], lq[8];
    const int t = threadIdx.x;
    const float* xp = x + (size_t)blockIdx.x * EMB;
    const float v0 = xp[t], v1 = xp[t + 256], v2 = xp[t + 512];
    float s  = v0 + v1 + v2;
    float sq = v0 * v0 + v1 * v1 + v2 * v2;
    #pragma unroll
    for (int m = 1; m < 32; m <<= 1) {
        s  += __shfl_xor(s,  m, 32);
        sq += __shfl_xor(sq, m, 32);
    }
    if ((t & 31) == 0) { ls[t >> 5] = s; lq[t >> 5] = sq; }
    __syncthreads();
    float ts = 0.f, tq = 0.f;
    #pragma unroll
    for (int i = 0; i < 8; ++i) { ts += ls[i]; tq += lq[i]; }
    const float mean = ts * (1.0f / EMB);
    const float var  = tq * (1.0f / EMB) - mean * mean;
    const float rstd = rsqrtf(var + 1e-6f);
    const size_t base = (size_t)blockIdx.x * EMB;
    const float vv[3] = {v0, v1, v2};
    #pragma unroll
    for (int i = 0; i < 3; ++i) {
        const int c = t + i * 256;
        const float y = (vv[i] - mean) * rstd * gam[c] + bet[c];
        if (ybf) ybf[base + c] = f2bf(y);
        else     yf[base + c]  = y;
    }
}

// ---------------------------------------------------------------------------
// Weight convert + transpose: W fp32 [K][N] -> Wt bf16 [N][K]
// ---------------------------------------------------------------------------
__global__ __launch_bounds__(256) void wt_kernel(
    const float* __restrict__ in, unsigned short* __restrict__ out, int K, int N)
{
    const int idx = blockIdx.x * 256 + threadIdx.x;
    if (idx < K * N) {
        const int k = idx / N, n = idx % N;
        out[(size_t)n * K + k] = f2bf(in[idx]);
    }
}

// ---------------------------------------------------------------------------
// Split qkv_bf16 [tok][3E] -> Q,K [b,h,n,d] and V transposed [b,h,d,n]
// ---------------------------------------------------------------------------
__global__ __launch_bounds__(256) void qkv_split_kernel(
    const unsigned short* __restrict__ qkv, unsigned short* __restrict__ Qb,
    unsigned short* __restrict__ Kb, unsigned short* __restrict__ Vt)
{
    const int idx = blockIdx.x * 256 + threadIdx.x;   // < TOKENS*3E
    const int tok = idx / (3 * EMB), c = idx % (3 * EMB);
    const int b = tok >> 11, n = tok & (NSEQ - 1);
    const int sec = c / EMB, hd = c % EMB;
    const int head = hd >> 6, d = hd & 63;
    const unsigned short v = qkv[idx];
    const int bh = b * NHEAD + head;
    if      (sec == 0) Qb[((size_t)bh * NSEQ + n) * DH + d] = v;
    else if (sec == 1) Kb[((size_t)bh * NSEQ + n) * DH + d] = v;
    else               Vt[((size_t)bh * DH + d) * NSEQ + n] = v;
}

// ---------------------------------------------------------------------------
// Flash attention: one wave handles 32 query rows of one (b,head).
// S = Q@K^T (WMMA), online softmax w/ ALiBi+causal (shuffle reductions),
// P transposed through per-wave LDS tile, O += P@V (WMMA, V pre-transposed).
// ---------------------------------------------------------------------------
struct Slopes { float s[NHEAD]; };

__global__ __launch_bounds__(256) void attn_kernel(
    const unsigned short* __restrict__ Qb, const unsigned short* __restrict__ Kb,
    const unsigned short* __restrict__ Vt, unsigned short* __restrict__ Ob,
    Slopes sl)
{
    __shared__ __align__(16) unsigned short lP[8][16][40];  // per-wave P tile, pad 8

    const int wib  = threadIdx.x >> 5;
    const int lane = threadIdx.x & 31;
    const int ln   = lane & 15, g = lane >> 4;
    const int wid  = blockIdx.x * 8 + wib;        // [0, B*H*N/32)
    const int qb   = wid & 63;                    // N/32 = 64
    const int bh   = wid >> 6;                    // [0, 24)
    const int head = bh % NHEAD;
    const float slope = sl.s[head];
    const int q0 = qb * 32;
    const float scale = 0.125f;                   // Dh^-0.5

    const unsigned short* Qp = Qb + (size_t)bh * NSEQ * DH;
    const unsigned short* Kp = Kb + (size_t)bh * NSEQ * DH;
    const unsigned short* Vp = Vt + (size_t)bh * DH * NSEQ;

    // Q A-fragments: qa[mi][kc] covers rows q0+mi*16.., K-chunk kc*32
    Frag qa[2][2];
    #pragma unroll
    for (int mi = 0; mi < 2; ++mi)
        #pragma unroll
        for (int kc = 0; kc < 2; ++kc) {
            const unsigned short* p = Qp + (size_t)(q0 + mi * 16 + ln) * DH + kc * 32;
            qa[mi][kc].q[0] = *(const uint4*)(p + 8 * g);
            qa[mi][kc].q[1] = *(const uint4*)(p + 16 + 8 * g);
        }

    float mrun[2][8], lrun[2][8];
    v8f o[2][4];
    #pragma unroll
    for (int mi = 0; mi < 2; ++mi) {
        #pragma unroll
        for (int r = 0; r < 8; ++r) { mrun[mi][r] = -3.0e38f; lrun[mi][r] = 0.0f; }
        #pragma unroll
        for (int nc = 0; nc < 4; ++nc) o[mi][nc] = v8f_zero();
    }

    const int jend = q0 >> 5;   // inclusive (causal)
    for (int jb = 0; jb <= jend; ++jb) {
        const int kb0 = jb * 32;

        Frag kf[2][2];   // B-frags of K^T for S: key-subtile tt, K-chunk kc
        #pragma unroll
        for (int tt = 0; tt < 2; ++tt)
            #pragma unroll
            for (int kc = 0; kc < 2; ++kc) {
                const unsigned short* p = Kp + (size_t)(kb0 + tt * 16 + ln) * DH + kc * 32;
                kf[tt][kc].q[0] = *(const uint4*)(p + 8 * g);
                kf[tt][kc].q[1] = *(const uint4*)(p + 16 + 8 * g);
            }
        Frag vf[4];      // B-frags of V (32 keys x 16 d), from transposed V
        #pragma unroll
        for (int nc = 0; nc < 4; ++nc) {
            const unsigned short* p = Vp + (size_t)(nc * 16 + ln) * NSEQ + kb0 + 16 * g;
            vf[nc].q[0] = *(const uint4*)(p);
            vf[nc].q[1] = *(const uint4*)(p + 8);
        }

        #pragma unroll
        for (int mi = 0; mi < 2; ++mi) {
            v8f s0 = v8f_zero(), s1 = v8f_zero();
            s0 = wmma_bf16(qa[mi][0], kf[0][0], s0);
            s0 = wmma_bf16(qa[mi][1], kf[0][1], s0);
            s1 = wmma_bf16(qa[mi][0], kf[1][0], s1);
            s1 = wmma_bf16(qa[mi][1], kf[1][1], s1);

            float alpha[8];
            #pragma unroll
            for (int r = 0; r < 8; ++r) {
                const int rowq = q0 + mi * 16 + g * 8 + r;
                const int key0 = kb0 + ln, key1 = key0 + 16;
                float a0 = s0[r] * scale + slope * (float)key0;
                float a1 = s1[r] * scale + slope * (float)key1;
                if (key0 > rowq) a0 = -3.0e38f;
                if (key1 > rowq) a1 = -3.0e38f;
                float mx = fmaxf(a0, a1);
                #pragma unroll
                for (int mm = 1; mm < 16; mm <<= 1) mx = fmaxf(mx, __shfl_xor(mx, mm, 32));
                const float mnew = fmaxf(mrun[mi][r], mx);
                const float p0 = __expf(a0 - mnew), p1 = __expf(a1 - mnew);
                float sum = p0 + p1;
                #pragma unroll
                for (int mm = 1; mm < 16; mm <<= 1) sum += __shfl_xor(sum, mm, 32);
                alpha[r]    = __expf(mrun[mi][r] - mnew);
                mrun[mi][r] = mnew;
                lrun[mi][r] = lrun[mi][r] * alpha[r] + sum;
                lP[wib][g * 8 + r][ln]      = f2bf(p0);   // P transpose via LDS
                lP[wib][g * 8 + r][16 + ln] = f2bf(p1);
            }
            #pragma unroll
            for (int nc = 0; nc < 4; ++nc)
                #pragma unroll
                for (int r = 0; r < 8; ++r)
                    o[mi][nc][r] *= alpha[r];

            Frag pf;   // P as A-fragment (16 rows x 32 keys)
            pf.q[0] = *(const uint4*)&lP[wib][ln][16 * g];
            pf.q[1] = *(const uint4*)&lP[wib][ln][16 * g + 8];
            #pragma unroll
            for (int nc = 0; nc < 4; ++nc)
                o[mi][nc] = wmma_bf16(pf, vf[nc], o[mi][nc]);
        }
    }

    // normalize and scatter to [b, n, E] bf16
    const int b = bh / NHEAD;
    #pragma unroll
    for (int mi = 0; mi < 2; ++mi) {
        float inv[8];
        #pragma unroll
        for (int r = 0; r < 8; ++r) inv[r] = __builtin_amdgcn_rcpf(lrun[mi][r]);
        #pragma unroll
        for (int nc = 0; nc < 4; ++nc) {
            const int col = head * 64 + nc * 16 + ln;
            #pragma unroll
            for (int r = 0; r < 8; ++r) {
                const int tok = (b << 11) + q0 + mi * 16 + g * 8 + r;
                Ob[(size_t)tok * EMB + col] = f2bf(o[mi][nc][r] * inv[r]);
            }
        }
    }
}

// ---------------------------------------------------------------------------
// Host side
// ---------------------------------------------------------------------------
static void compute_slopes(float* s) {
    const int n = NHEAD;
    int c = 1;
    while ((c << 1) <= n) c <<= 1;                                   // 8
    const double st = pow(2.0, -pow(2.0, -(log2((double)c) - 3.0)));
    for (int i = 0; i < c; ++i) s[i] = (float)pow(st, i + 1);
    if (c < n) {
        const double st2 = pow(2.0, -pow(2.0, -(log2((double)(2 * c)) - 3.0)));
        for (int i = 0; i < n - c; ++i) s[c + i] = (float)pow(st2, 2 * i + 1);
    }
}

extern "C" void kernel_launch(void* const* d_in, const int* in_sizes, int n_in,
                              void* d_out, int out_size, void* d_ws, size_t ws_size,
                              hipStream_t stream) {
    (void)in_sizes; (void)n_in; (void)out_size; (void)ws_size;
    const float* x    = (const float*)d_in[0];
    const float* wqkv = (const float*)d_in[1];
    const float* bqkv = (const float*)d_in[2];
    const float* wo   = (const float*)d_in[3];
    const float* bo   = (const float*)d_in[4];
    const float* ln1s = (const float*)d_in[5];
    const float* ln1b = (const float*)d_in[6];
    const float* ln2s = (const float*)d_in[7];
    const float* ln2b = (const float*)d_in[8];
    const float* w1   = (const float*)d_in[9];
    const float* w2   = (const float*)d_in[10];
    const float* lnfs = (const float*)d_in[11];
    const float* lnfb = (const float*)d_in[12];

    char* ws = (char*)d_ws;
    size_t off = 0;
    auto alloc = [&](size_t bytes) -> void* {
        void* p = ws + off;
        off += (bytes + 255) & ~(size_t)255;
        return p;
    };
    float*          h     = (float*)alloc((size_t)TOKENS * EMB * 4);
    unsigned short* qkvb  = (unsigned short*)alloc((size_t)TOKENS * 3 * EMB * 2);
    unsigned short* ybf   = (unsigned short*)alloc((size_t)TOKENS * EMB * 2);
    unsigned short* Qb    = (unsigned short*)alloc((size_t)TOKENS * EMB * 2);
    unsigned short* Kbuf  = (unsigned short*)alloc((size_t)TOKENS * EMB * 2);
    unsigned short* Vt    = (unsigned short*)alloc((size_t)TOKENS * EMB * 2);
    unsigned short* obf   = (unsigned short*)alloc((size_t)TOKENS * EMB * 2);
    unsigned short* actbf = (unsigned short*)alloc((size_t)TOKENS * 4 * EMB * 2);
    unsigned short* wbuf  = (unsigned short*)alloc((size_t)EMB * 4 * EMB * 2);

    Slopes sl;
    compute_slopes(sl.s);

    hipMemcpyAsync(h, x, (size_t)TOKENS * EMB * 4, hipMemcpyDeviceToDevice, stream);

    const dim3 blk(256);
    for (int l = 0; l < 6; ++l) {
        // LN1 -> bf16
        ln_kernel<<<TOKENS, blk, 0, stream>>>(h, ln1s + l * EMB, ln1b + l * EMB, ybf, nullptr);
        // QKV projection (bias, bf16 out)
        wt_kernel<<<(EMB * 3 * EMB) / 256, blk, 0, stream>>>(wqkv + (size_t)l * EMB * 3 * EMB, wbuf, EMB, 3 * EMB);
        gemm_bf16<true, false, false, true><<<dim3((3 * EMB) / 128, TOKENS / 128), blk, 0, stream>>>(
            ybf, wbuf, bqkv + (size_t)l * 3 * EMB, nullptr, nullptr, qkvb, TOKENS, 3 * EMB, EMB);
        qkv_split_kernel<<<(TOKENS * 3 * EMB) / 256, blk, 0, stream>>>(qkvb, Qb, Kbuf, Vt);
        // attention
        attn_kernel<<<(NBAT * NHEAD * (NSEQ / 32)) / 8, blk, 0, stream>>>(Qb, Kbuf, Vt, obf, sl);
        // output projection + residual (h updated in place)
        wt_kernel<<<(EMB * EMB) / 256, blk, 0, stream>>>(wo + (size_t)l * EMB * EMB, wbuf, EMB, EMB);
        gemm_bf16<true, true, false, false><<<dim3(EMB / 128, TOKENS / 128), blk, 0, stream>>>(
            obf, wbuf, bo + (size_t)l * EMB, h, h, nullptr, TOKENS, EMB, EMB);
        // LN2 -> bf16
        ln_kernel<<<TOKENS, blk, 0, stream>>>(h, ln2s + l * EMB, ln2b + l * EMB, ybf, nullptr);
        // MLP: gelu(y@W1)@W2 + residual
        wt_kernel<<<(EMB * 4 * EMB) / 256, blk, 0, stream>>>(w1 + (size_t)l * EMB * 4 * EMB, wbuf, EMB, 4 * EMB);
        gemm_bf16<false, false, true, true><<<dim3((4 * EMB) / 128, TOKENS / 128), blk, 0, stream>>>(
            ybf, wbuf, nullptr, nullptr, nullptr, actbf, TOKENS, 4 * EMB, EMB);
        wt_kernel<<<(4 * EMB * EMB) / 256, blk, 0, stream>>>(w2 + (size_t)l * 4 * EMB * EMB, wbuf, 4 * EMB, EMB);
        gemm_bf16<false, true, false, false><<<dim3(EMB / 128, TOKENS / 128), blk, 0, stream>>>(
            actbf, wbuf, nullptr, h, h, nullptr, TOKENS, EMB, 4 * EMB);
    }
    // final LN -> fp32 output
    ln_kernel<<<TOKENS, blk, 0, stream>>>(h, lnfs, lnfb, nullptr, (float*)d_out);
}
